// ResidualVQ_13125420056613
// MI455X (gfx1250) — matrix operand order repaired and checked by reading, not compiled, not run on hardware
//
#include <hip/hip_runtime.h>
#include <hip/hip_bf16.h>

typedef __attribute__((ext_vector_type(16))) __bf16 v16bf;
typedef __attribute__((ext_vector_type(8)))  float  v8f;

#define RVQ_N   64
#define RVQ_D   512
#define RVQ_T   256
#define RVQ_Q   6
#define RVQ_C   1024
#define RVQ_NT  (RVQ_N * RVQ_T)                   // 16384 rows
#define RVQ_NDT ((size_t)RVQ_N * RVQ_D * RVQ_T)   // 8388608
#define RVQ_EPS 1e-10f
#define RVQ_TILE_BYTES 16384                      // one B tile: 16 frags * 32 lanes * 32B

// ---------------------------------------------------------------------------
// Init: zero code-usage histograms (ws is poisoned; launch must be idempotent)
// ---------------------------------------------------------------------------
__global__ void rvq_init_kernel(unsigned int* __restrict__ hist) {
    int i = blockIdx.x * blockDim.x + threadIdx.x;
    if (i < RVQ_Q * RVQ_C) hist[i] = 0u;
}

// ---------------------------------------------------------------------------
// Tiled transpose: x (N, D, T) -> residual (N*T, D); zero quant accumulator.
// 32x32 tiles through LDS, coalesced both sides. grid (T/32, D/32, N).
// ---------------------------------------------------------------------------
__global__ void __launch_bounds__(256)
rvq_pack_x_kernel(const float* __restrict__ x,
                  float* __restrict__ residual,
                  float* __restrict__ quant) {
    __shared__ float tile[32][33];
    const int n  = blockIdx.z;
    const int d0 = blockIdx.y * 32;
    const int t0 = blockIdx.x * 32;
    const int tx = threadIdx.x & 31;
    const int ty = threadIdx.x >> 5;       // 0..7
#pragma unroll
    for (int i = 0; i < 4; ++i) {          // read coalesced in t
        int d = d0 + ty + i * 8;
        tile[ty + i * 8][tx] = x[((size_t)n * RVQ_D + d) * RVQ_T + t0 + tx];
    }
    __syncthreads();
#pragma unroll
    for (int i = 0; i < 4; ++i) {          // write coalesced in d
        int t = t0 + ty + i * 8;
        size_t o = ((size_t)n * RVQ_T + t) * RVQ_D + d0 + tx;
        residual[o] = tile[tx][ty + i * 8];
        quant[o]    = 0.0f;
    }
}

// ---------------------------------------------------------------------------
// Pack codebooks (Q, C, D) fp32 -> bf16 B-fragment layout.
// B 32x16 (wave32), contiguous k order used here (A matches): lane l holds column N = l%16,
// K = (l/16)*16 + e  (e = 0..15, packed pairs per VGPR).
// Flat: ((((q*64 + ct)*16 + ks)*32 + lane)*16 + e)
// ---------------------------------------------------------------------------
__global__ void rvq_pack_cb_kernel(const float* __restrict__ cb,
                                   __bf16* __restrict__ cbB) {
    size_t i = (size_t)blockIdx.x * blockDim.x + threadIdx.x;
    if (i >= (size_t)RVQ_Q * RVQ_C * RVQ_D) return;
    size_t e    = i % 16;  size_t r = i / 16;
    size_t lane = r % 32;  r /= 32;
    size_t ks   = r % 16;  r /= 16;
    size_t ct   = r % 64;  size_t q = r / 64;
    size_t c = ct * 16 + (lane & 15);
    size_t d = ks * 32 + (lane >> 4) * 16 + e;
    cbB[i] = (__bf16)cb[(q * RVQ_C + c) * RVQ_D + d];
}

// ---------------------------------------------------------------------------
// Per-code squared L2 norms, fp32.
// ---------------------------------------------------------------------------
__global__ void rvq_cbnorm_kernel(const float* __restrict__ cb,
                                  float* __restrict__ cbnorm) {
    int i = blockIdx.x * blockDim.x + threadIdx.x;  // over Q*C
    if (i >= RVQ_Q * RVQ_C) return;
    const float* p = cb + (size_t)i * RVQ_D;
    float s = 0.0f;
    for (int d = 0; d < RVQ_D; ++d) { float v = p[d]; s += v * v; }
    cbnorm[i] = s;
}

// ---------------------------------------------------------------------------
// Fused WMMA distance + argmin with async-LDS staged B tiles.
// dist(m,c) = ||cb_c||^2 - 2 <x_m, cb_c>    (||x||^2 constant per row: drop)
// - A (K=512, 16 frags) resident in VGPRs per wave (one 16-row tile each).
// - B tile (16 KB) staged ONCE per block into LDS via
//   global_load_async_to_lds_b128 (ASYNCcnt), double-buffered: tile ct+1
//   streams in while the 16 WMMAs consume tile ct via ds_load_b128.
// - Dual accumulators break the WMMA RAW chain.
// ---------------------------------------------------------------------------
__global__ void __launch_bounds__(128)
rvq_argmin_kernel(const float* __restrict__ residual,
                  const __bf16* __restrict__ cbB_stage,
                  const float* __restrict__ cbnorm_stage,
                  int* __restrict__ idx_out) {
    __shared__ __bf16 ldsb[2][RVQ_TILE_BYTES / 2];   // two 16 KB B tiles

    const int tid   = threadIdx.x;
    const int lane  = tid & 31;
    const int wave  = tid >> 5;
    const int tile  = blockIdx.x * 4 + wave;   // 0..1023
    const int rowb  = tile * 16;
    const int m     = lane & 15;
    const int khalf = lane >> 4;

    // ---- A fragments: lane holds row M = lane%16; e -> K = (l/16)*16+e
    const float* rowp = residual + (size_t)(rowb + m) * RVQ_D;
    v16bf a[16];
#pragma unroll
    for (int ks = 0; ks < 16; ++ks) {
        const int d0 = ks * 32 + khalf * 16;
        const float4* p0 = (const float4*)(rowp + d0);
        const float4* p1 = (const float4*)(rowp + d0 + 8);
        float4 f0 = p0[0], f1 = p0[1], f2 = p1[0], f3 = p1[1];
        v16bf f;
        f[0]  = (__bf16)f0.x; f[1]  = (__bf16)f0.y; f[2]  = (__bf16)f0.z; f[3]  = (__bf16)f0.w;
        f[4]  = (__bf16)f1.x; f[5]  = (__bf16)f1.y; f[6]  = (__bf16)f1.z; f[7]  = (__bf16)f1.w;
        f[8]  = (__bf16)f2.x; f[9]  = (__bf16)f2.y; f[10] = (__bf16)f2.z; f[11] = (__bf16)f2.w;
        f[12] = (__bf16)f3.x; f[13] = (__bf16)f3.y; f[14] = (__bf16)f3.z; f[15] = (__bf16)f3.w;
        a[ks] = f;
    }

    // ---- async staging addresses: 128 threads x 8 b128 copies = 16 KB tile
    const unsigned tb   = (unsigned)tid * 16u;
    const unsigned lds0 = (unsigned)(size_t)(&ldsb[0][0]) + tb;
    const unsigned lds1 = (unsigned)(size_t)(&ldsb[1][0]) + tb;
    unsigned long long g = (unsigned long long)(size_t)cbB_stage + tb;

    // prologue: stage tile 0 into buffer 0
    {
        unsigned lo = lds0;
        unsigned long long gg = g;
#pragma unroll
        for (int j = 0; j < 8; ++j) {
            asm volatile("global_load_async_to_lds_b128 %0, %1, off"
                         :: "v"(lo), "v"(gg) : "memory");
            lo += 2048u; gg += 2048ull;
        }
        g += (unsigned long long)RVQ_TILE_BYTES;
    }
    asm volatile("s_wait_asynccnt 0x0" ::: "memory");
    __syncthreads();

    const v16bf* f0 = (const v16bf*)(&ldsb[0][0]) + lane;
    const v16bf* f1 = (const v16bf*)(&ldsb[1][0]) + lane;

    float minv[8];
    int   mini[8];
#pragma unroll
    for (int r = 0; r < 8; ++r) { minv[r] = 3.4e38f; mini[r] = 0; }

    for (int ct = 0; ct < 64; ++ct) {
        // stream tile ct+1 into the other buffer while we compute on tile ct
        if (ct < 63) {
            unsigned lo = (ct & 1) ? lds0 : lds1;
            unsigned long long gg = g;
#pragma unroll
            for (int j = 0; j < 8; ++j) {
                asm volatile("global_load_async_to_lds_b128 %0, %1, off"
                             :: "v"(lo), "v"(gg) : "memory");
                lo += 2048u; gg += 2048ull;
            }
            g += (unsigned long long)RVQ_TILE_BYTES;
        }

        const v16bf* curf = (ct & 1) ? f1 : f0;
        v8f acc0 = {}, acc1 = {};
#pragma unroll
        for (int k = 0; k < 16; k += 2) {
            v16bf bA = curf[(size_t)k * 32];
            v16bf bB = curf[(size_t)(k + 1) * 32];
            acc0 = __builtin_amdgcn_wmma_f32_16x16x32_bf16(
                false, a[k],     false, bA, (short)0, acc0, false, false);
            acc1 = __builtin_amdgcn_wmma_f32_16x16x32_bf16(
                false, a[k + 1], false, bB, (short)0, acc1, false, false);
        }

        const int col = ct * 16 + m;
        const float cn = cbnorm_stage[col];
#pragma unroll
        for (int r = 0; r < 8; ++r) {
            float dist = fmaf(-2.0f, acc0[r] + acc1[r], cn);
            if (dist < minv[r]) { minv[r] = dist; mini[r] = col; }
        }

        if (ct < 63)
            asm volatile("s_wait_asynccnt 0x0" ::: "memory");
        __syncthreads();
    }

    // min-reduce across the 16 lanes sharing a row set (xor 1,2,4,8)
#pragma unroll
    for (int off = 1; off < 16; off <<= 1) {
#pragma unroll
        for (int r = 0; r < 8; ++r) {
            float ov = __shfl_xor(minv[r], off, 32);
            int   oi = __shfl_xor(mini[r], off, 32);
            if (ov < minv[r] || (ov == minv[r] && oi < mini[r])) {
                minv[r] = ov; mini[r] = oi;
            }
        }
    }
    if (m == 0) {
#pragma unroll
        for (int r = 0; r < 8; ++r)
            idx_out[rowb + khalf * 8 + r] = mini[r];
    }
}

// ---------------------------------------------------------------------------
// Gather chosen code, update residual & quant, per-row squared-residual
// partial (fixed order -> deterministic), integer histogram.
// commit_loss == mean(new_residual^2).
// ---------------------------------------------------------------------------
__global__ void __launch_bounds__(256)
rvq_update_kernel(float* __restrict__ residual,
                  float* __restrict__ quant,
                  const float* __restrict__ cb_stage,
                  const int* __restrict__ idx_stage,
                  unsigned int* __restrict__ hist_stage,
                  float* __restrict__ partials) {
    const int row = blockIdx.x;                // 0..NT-1
    const int idx = idx_stage[row];
    const float* code = cb_stage + (size_t)idx * RVQ_D;
    float* r = residual + (size_t)row * RVQ_D;
    float* q = quant    + (size_t)row * RVQ_D;

    float s = 0.0f;
    for (int d = threadIdx.x; d < RVQ_D; d += 256) {
        float xd = code[d];
        float rn = r[d] - xd;
        r[d] = rn;
        q[d] += xd;
        s += rn * rn;
    }
    __shared__ float red[256];
    red[threadIdx.x] = s;
    __syncthreads();
    for (int st = 128; st > 0; st >>= 1) {
        if (threadIdx.x < st) red[threadIdx.x] += red[threadIdx.x + st];
        __syncthreads();
    }
    if (threadIdx.x == 0) {
        partials[row] = red[0];
        atomicAdd(&hist_stage[idx], 1u);
    }
}

// ---------------------------------------------------------------------------
// Per-stage scalars: commit loss + perplexity (deterministic tree reduce).
// ---------------------------------------------------------------------------
__global__ void __launch_bounds__(1024)
rvq_stats_kernel(const float* __restrict__ partials,
                 const unsigned int* __restrict__ hist_stage,
                 float* __restrict__ loss_q,
                 float* __restrict__ perp_q) {
    __shared__ float s1[1024];
    __shared__ float s2[1024];
    const int t = threadIdx.x;
    float a = 0.0f;
    for (int i = t; i < RVQ_NT; i += 1024) a += partials[i];
    float p = (float)hist_stage[t] * (1.0f / (float)RVQ_NT);
    float e = p * logf(p + RVQ_EPS);
    s1[t] = a; s2[t] = e;
    __syncthreads();
    for (int st = 512; st > 0; st >>= 1) {
        if (t < st) { s1[t] += s1[t + st]; s2[t] += s2[t + st]; }
        __syncthreads();
    }
    if (t == 0) {
        loss_q[0] = s1[0] / ((float)RVQ_NT * (float)RVQ_D);
        perp_q[0] = expf(-s2[0]);
    }
}

// ---------------------------------------------------------------------------
// Output 1: tiled transpose quant (N*T, D) -> out (N, D, T), coalesced.
// ---------------------------------------------------------------------------
__global__ void __launch_bounds__(256)
rvq_out_quant_kernel(const float* __restrict__ quant, float* __restrict__ out) {
    __shared__ float tile[32][33];
    const int n  = blockIdx.z;
    const int t0 = blockIdx.y * 32;
    const int d0 = blockIdx.x * 32;
    const int tx = threadIdx.x & 31;
    const int ty = threadIdx.x >> 5;
#pragma unroll
    for (int i = 0; i < 4; ++i) {          // read coalesced in d
        int t = t0 + ty + i * 8;
        tile[ty + i * 8][tx] = quant[((size_t)n * RVQ_T + t) * RVQ_D + d0 + tx];
    }
    __syncthreads();
#pragma unroll
    for (int i = 0; i < 4; ++i) {          // write coalesced in t
        int d = d0 + ty + i * 8;
        out[((size_t)n * RVQ_D + d) * RVQ_T + t0 + tx] = tile[tx][ty + i * 8];
    }
}

// ---------------------------------------------------------------------------
// Output 2: indices (N, T, Q) as float + two scalar means.
// ---------------------------------------------------------------------------
__global__ void rvq_out_tail_kernel(const int* __restrict__ idx_ws,
                                    const float* __restrict__ loss_q,
                                    const float* __restrict__ perp_q,
                                    float* __restrict__ out) {
    size_t i = (size_t)blockIdx.x * blockDim.x + threadIdx.x;
    const size_t NIDX = (size_t)RVQ_NT * RVQ_Q;
    if (i < NIDX) {
        size_t q  = i % RVQ_Q;
        size_t nt = i / RVQ_Q;
        out[RVQ_NDT + i] = (float)idx_ws[q * RVQ_NT + nt];
    } else if (i == NIDX) {
        float s = 0.0f;
        for (int q = 0; q < RVQ_Q; ++q) s += loss_q[q];
        out[RVQ_NDT + i] = s / (float)RVQ_Q;
    } else if (i == NIDX + 1) {
        float s = 0.0f;
        for (int q = 0; q < RVQ_Q; ++q) s += perp_q[q];
        out[RVQ_NDT + i] = s / (float)RVQ_Q;
    }
}

// ---------------------------------------------------------------------------
extern "C" void kernel_launch(void* const* d_in, const int* in_sizes, int n_in,
                              void* d_out, int out_size, void* d_ws, size_t ws_size,
                              hipStream_t stream) {
    (void)in_sizes; (void)n_in; (void)out_size; (void)ws_size;
    const float* x  = (const float*)d_in[0];   // (64, 512, 256)
    const float* cb = (const float*)d_in[1];   // (6, 1024, 512)
    float* out = (float*)d_out;

    char* ws = (char*)d_ws;
    size_t off = 0;
    auto alloc = [&](size_t bytes) -> char* {
        char* p = ws + off;
        off += (bytes + 255) & ~(size_t)255;
        return p;
    };
    float*        residual = (float*)alloc((size_t)RVQ_NT * RVQ_D * 4);
    float*        quant    = (float*)alloc((size_t)RVQ_NT * RVQ_D * 4);
    __bf16*       cbB      = (__bf16*)alloc((size_t)RVQ_Q * RVQ_C * RVQ_D * 2);
    float*        cbnorm   = (float*)alloc((size_t)RVQ_Q * RVQ_C * 4);
    int*          idxws    = (int*)alloc((size_t)RVQ_Q * RVQ_NT * 4);
    unsigned int* hist     = (unsigned int*)alloc((size_t)RVQ_Q * RVQ_C * 4);
    float*        partials = (float*)alloc((size_t)RVQ_NT * 4);
    float*        lossq    = (float*)alloc(RVQ_Q * 4);
    float*        perpq    = (float*)alloc(RVQ_Q * 4);

    rvq_init_kernel<<<(RVQ_Q * RVQ_C + 255) / 256, 256, 0, stream>>>(hist);
    rvq_pack_x_kernel<<<dim3(RVQ_T / 32, RVQ_D / 32, RVQ_N), 256, 0, stream>>>(
        x, residual, quant);
    rvq_pack_cb_kernel<<<(unsigned)(((size_t)RVQ_Q * RVQ_C * RVQ_D + 255) / 256), 256, 0, stream>>>(
        cb, cbB);
    rvq_cbnorm_kernel<<<(RVQ_Q * RVQ_C + 255) / 256, 256, 0, stream>>>(cb, cbnorm);

    for (int q = 0; q < RVQ_Q; ++q) {
        rvq_argmin_kernel<<<RVQ_NT / 16 / 4, 128, 0, stream>>>(
            residual,
            cbB + (size_t)q * RVQ_C * RVQ_D,
            cbnorm + (size_t)q * RVQ_C,
            idxws + (size_t)q * RVQ_NT);
        rvq_update_kernel<<<RVQ_NT, 256, 0, stream>>>(
            residual, quant,
            cb + (size_t)q * RVQ_C * RVQ_D,
            idxws + (size_t)q * RVQ_NT,
            hist + (size_t)q * RVQ_C,
            partials);
        rvq_stats_kernel<<<1, 1024, 0, stream>>>(partials,
                                                 hist + (size_t)q * RVQ_C,
                                                 lossq + q, perpq + q);
    }

    rvq_out_quant_kernel<<<dim3(RVQ_D / 32, RVQ_T / 32, RVQ_N), 256, 0, stream>>>(quant, out);
    const size_t tail = (size_t)RVQ_NT * RVQ_Q + 2;
    rvq_out_tail_kernel<<<(unsigned)((tail + 255) / 256), 256, 0, stream>>>(
        idxws, lossq, perpq, out);
}
